// DattaBotModel_50903952392669
// MI455X (gfx1250) — matrix operand order repaired and checked by reading, not compiled, not run on hardware
//
#include <hip/hip_runtime.h>
#include <cmath>

// ---------------------------------------------------------------------------
// Types / constants
// ---------------------------------------------------------------------------
typedef unsigned short u16b; // raw bf16 storage
typedef __attribute__((ext_vector_type(16))) __bf16 v16bf;
typedef __attribute__((ext_vector_type(8)))  float  v8f;
typedef __attribute__((ext_vector_type(4)))  unsigned int u32x4;
typedef __attribute__((ext_vector_type(8)))  int i32x8;
typedef __attribute__((ext_vector_type(4)))  int i32x4;

#define Bc   4
#define Sc   1024
#define Dc   1024
#define Hc   16
#define DHc  64
#define FFc  4096
#define Lc   2
#define ROWS (Bc * Sc) // 4096

#if __has_builtin(__builtin_amdgcn_tensor_load_to_lds)
#define HAVE_TDM 1
#else
#define HAVE_TDM 0
#endif

union Frag {
  uint4 u4[2];
  v16bf v;
  u16b  h[16];
};

__device__ __forceinline__ u16b f2bf(float f) {
  unsigned u = __float_as_uint(f);
  unsigned r = u + 0x7fffu + ((u >> 16) & 1u); // round-to-nearest-even
  return (u16b)(r >> 16);
}

__device__ __forceinline__ void wait_ds0() {
#if __has_builtin(__builtin_amdgcn_s_wait_dscnt)
  __builtin_amdgcn_s_wait_dscnt(0);
#else
  asm volatile("s_wait_dscnt 0x0" ::: "memory");
#endif
}

__device__ __forceinline__ void wait_tensor0() {
#if __has_builtin(__builtin_amdgcn_s_wait_tensorcnt)
  __builtin_amdgcn_s_wait_tensorcnt(0);
#else
  asm volatile("s_wait_tensorcnt 0x0" ::: "memory");
#endif
}

// Low 32 bits of a flat shared pointer == LDS byte offset (shared aperture
// encoding: addr = {SHARED_BASE[63:32], lds_offset32}).
__device__ __forceinline__ unsigned lds_off(const void* p) {
  return (unsigned)(size_t)p;
}

#if HAVE_TDM
// Tensor Data Mover: 2-D tile load (bf16 elements) global -> LDS with LDS row
// padding. D# packing per CDNA5 ISA.
//   tile_d0  : contiguous elements per row   (K extent)
//   tile_d1  : number of rows
//   stride   : tensor dim0 stride in elements
//   pad_iv   : pad interval code (pad after (1<<code)*2 DWORDs)
//   pad_amt  : pad amount code   (code+1 DWORDs inserted)
__device__ __forceinline__ void tdm_load_2d(unsigned dst_lds, const void* gsrc,
                                            unsigned tensor_d0, unsigned tensor_d1,
                                            unsigned tile_d0, unsigned tile_d1,
                                            unsigned stride,
                                            unsigned pad_iv, unsigned pad_amt) {
  size_t ga = (size_t)gsrc;
  u32x4 g0;
  g0[0] = 1u;                                   // count=1 (valid), user mode
  g0[1] = dst_lds;                              // lds_addr [63:32]
  g0[2] = (unsigned)ga;                         // global_addr[31:0]
  g0[3] = (unsigned)((ga >> 32) & 0x1FFFFFFu)   // global_addr[56:32]
          | (2u << 30);                         // type=2 ("image")
  i32x8 g1;
  g1[0] = (int)((1u << 16)                      // data_size = 2 bytes
                | (1u << 20)                    // pad_enable
                | (pad_iv << 22)                // pad_interval
                | (pad_amt << 25));             // pad_amount
  g1[1] = (int)((tensor_d0 & 0xffffu) << 16);   // tensor_dim0[15:0] at bit 48
  g1[2] = (int)((tensor_d0 >> 16) | ((tensor_d1 & 0xffffu) << 16));
  g1[3] = (int)((tensor_d1 >> 16) | (tile_d0 << 16));   // tile_dim0 [127:112]
  g1[4] = (int)(tile_d1 & 0xffffu);                      // tile_dim1, tile_dim2=0
  g1[5] = (int)stride;                                   // dim0_stride[31:0]
  g1[6] = 0;                                             // stride hi, dim1_stride lo
  g1[7] = 0;
  i32x4 gz4 = {0, 0, 0, 0};                              // groups 2/3: zero (2-D)
  i32x8 gz8 = {0, 0, 0, 0, 0, 0, 0, 0};                  // trailing group: zero
  // 6-arg form (clang-23 / therock-10.0 toolchain)
  __builtin_amdgcn_tensor_load_to_lds(g0, g1, gz4, gz4, gz8, 0);
}
#endif

// ---------------------------------------------------------------------------
// Embedding + positional encoding: x[r,d] = emb[tok[r],d] + sin(d)
// ---------------------------------------------------------------------------
__global__ __launch_bounds__(256) void embed_kernel(const int* __restrict__ tok,
                                                    const float* __restrict__ emb,
                                                    float* __restrict__ x) {
  int i = blockIdx.x * 256 + threadIdx.x;     // 0 .. ROWS*Dc-1
  int row = i >> 10;
  int d = i & (Dc - 1);
  x[i] = emb[(size_t)tok[row] * Dc + d] + sinf((float)d);
}

// f32 -> bf16 convert (row-major activations)
__global__ __launch_bounds__(256) void convert_kernel(const float* __restrict__ src,
                                                      u16b* __restrict__ dst) {
  int i = blockIdx.x * 256 + threadIdx.x;
  dst[i] = f2bf(src[i]);
}

// Pack per-head weight [H, D, DH] -> bf16 [N=H*DH][K=D] (column-major GEMM B)
__global__ __launch_bounds__(256) void pack_head_kernel(const float* __restrict__ src,
                                                        u16b* __restrict__ dst) {
  int i = blockIdx.x * 256 + threadIdx.x;     // n*K + k, K = Dc
  int k = i & (Dc - 1);
  int n = i >> 10;
  dst[i] = f2bf(src[((size_t)(n >> 6) << 16) + (k << 6) + (n & 63)]);
}

// Pack plain weight [K, N] -> bf16 [N][K]
__global__ __launch_bounds__(256) void pack_plain_kernel(const float* __restrict__ src,
                                                         u16b* __restrict__ dst,
                                                         int K, int N) {
  int i = blockIdx.x * 256 + threadIdx.x;     // n*K + k
  int k = i % K;
  int n = i / K;
  dst[i] = f2bf(src[(size_t)k * N + n]);
}

// ---------------------------------------------------------------------------
// Tiled bf16 WMMA GEMM: C[M,N] = A[M,K] * B[K,N] (+bias, epilogue variants)
// A row-major bf16, Bp packed [N][K] bf16. Block = 256 thr (8 waves),
// tile 128x128, K-step 32, double-buffered LDS. Tiles are streamed by the
// Tensor Data Mover (wave 0 posts descriptors, s_wait_tensorcnt completes)
// overlapping DMA with the WMMA pipeline; cooperative-load fallback otherwise.
// ---------------------------------------------------------------------------
enum { EPI_BF16 = 0, EPI_RELU = 1, EPI_RESID = 2, EPI_SCALE = 3 };

template <int EPI>
__global__ __launch_bounds__(256) void gemm_kernel(const u16b* __restrict__ A,
                                                   const u16b* __restrict__ Bp,
                                                   const float* __restrict__ bias,
                                                   void* __restrict__ outp,
                                                   int N, int K, float scale) {
  __shared__ u16b lA[2][128][40]; // +8 pad: 80B row stride, 16B aligned
  __shared__ u16b lB[2][128][40];

  const int t = threadIdx.x;
  const int lane = t & 31, w = t >> 5;
  const int llo = lane & 15, lhi = lane >> 4;
  const int wm = w >> 1, wn = w & 1;
  const int m0 = blockIdx.y * 128;
  const int n0 = blockIdx.x * 128;

  v8f acc[2][4] = {};
  const int kSteps = K >> 5;

#if HAVE_TDM
  // prologue: DMA tile 0 into buffer 0
  if (w == 0) {
    tdm_load_2d(lds_off(&lA[0][0][0]), A + (size_t)m0 * K,
                (unsigned)K, (unsigned)ROWS, 32u, 128u, (unsigned)K, 3u, 3u);
    tdm_load_2d(lds_off(&lB[0][0][0]), Bp + (size_t)n0 * K,
                (unsigned)K, (unsigned)N, 32u, 128u, (unsigned)K, 3u, 3u);
    wait_tensor0();
  }
  __syncthreads();
#endif

  for (int ks = 0; ks < kSteps; ++ks) {
    const int buf = ks & 1;
#if HAVE_TDM
    // post DMA for the next K-tile while this one is consumed
    if (w == 0 && ks + 1 < kSteps) {
      const int k0n = (ks + 1) << 5;
      tdm_load_2d(lds_off(&lA[buf ^ 1][0][0]), A + (size_t)m0 * K + k0n,
                  (unsigned)K, (unsigned)ROWS, 32u, 128u, (unsigned)K, 3u, 3u);
      tdm_load_2d(lds_off(&lB[buf ^ 1][0][0]), Bp + (size_t)n0 * K + k0n,
                  (unsigned)K, (unsigned)N, 32u, 128u, (unsigned)K, 3u, 3u);
    }
#else
    {
      const int k0 = ks << 5;
#pragma unroll
      for (int c = t; c < 512; c += 256) {
        int row = c >> 2, q = c & 3;
        *(uint4*)&lA[buf][row][q * 8] =
            *(const uint4*)(A + (size_t)(m0 + row) * K + k0 + q * 8);
      }
#pragma unroll
      for (int c = t; c < 512; c += 256) {
        int row = c >> 2, q = c & 3;
        *(uint4*)&lB[buf][row][q * 8] =
            *(const uint4*)(Bp + (size_t)(n0 + row) * K + k0 + q * 8);
      }
      __syncthreads();
    }
#endif

    Frag af[2];
#pragma unroll
    for (int mt = 0; mt < 2; ++mt) {
      int row = wm * 32 + mt * 16 + llo;   // A: m = lane&15
      af[mt].u4[0] = *(const uint4*)&lA[buf][row][lhi * 8];      // k=lhi*8+0..7
      af[mt].u4[1] = *(const uint4*)&lA[buf][row][16 + lhi * 8]; // k=16+lhi*8+0..7
    }
#pragma unroll
    for (int nt = 0; nt < 4; ++nt) {
      int n = wn * 64 + nt * 16 + llo;     // B: n = lane&15
      Frag bfr;
      bfr.u4[0] = *(const uint4*)&lB[buf][n][lhi * 8];        // k=lhi*8+0..7
      bfr.u4[1] = *(const uint4*)&lB[buf][n][16 + lhi * 8];   // k=16+lhi*8+0..7
#pragma unroll
      for (int mt = 0; mt < 2; ++mt) {
        acc[mt][nt] = __builtin_amdgcn_wmma_f32_16x16x32_bf16(
            false, af[mt].v, false, bfr.v, (short)0, acc[mt][nt], false, false);
      }
    }

#if HAVE_TDM
    if (w == 0 && ks + 1 < kSteps) wait_tensor0();
    __syncthreads();
#else
    __syncthreads();
#endif
  }

  // Epilogue. C layout: VGPR r -> row (r + 8*lhi), col = lane&15 within tile.
#pragma unroll
  for (int mt = 0; mt < 2; ++mt) {
#pragma unroll
    for (int nt = 0; nt < 4; ++nt) {
      int gmb = m0 + wm * 32 + mt * 16 + (lhi << 3);
      int gn = n0 + wn * 64 + nt * 16 + llo;
      float bv = bias[gn];
#pragma unroll
      for (int r = 0; r < 8; ++r) {
        float val = acc[mt][nt][r] + bv;
        size_t idx = (size_t)(gmb + r) * N + gn;
        if (EPI == EPI_BF16) {
          ((u16b*)outp)[idx] = f2bf(val);
        } else if (EPI == EPI_SCALE) {
          ((u16b*)outp)[idx] = f2bf(val * scale);
        } else if (EPI == EPI_RELU) {
          ((u16b*)outp)[idx] = f2bf(fmaxf(val, 0.0f));
        } else { // EPI_RESID: accumulate into f32 residual stream
          ((float*)outp)[idx] += val;
        }
      }
    }
  }
}

// ---------------------------------------------------------------------------
// Flash attention: per wave one (b,h, 16-query tile); online softmax over
// 32-key tiles. q pre-scaled by 1/sqrt(DH). V tiles streamed per-wave by the
// TDM (posted before the score WMMAs, awaited just before P*V).
// ---------------------------------------------------------------------------
__global__ __launch_bounds__(256) void attention_kernel(const u16b* __restrict__ qb,
                                                        const u16b* __restrict__ kb,
                                                        const u16b* __restrict__ vb,
                                                        const int* __restrict__ mask,
                                                        u16b* __restrict__ ob) {
  __shared__ u16b pbuf[8][16][32];   // per-wave P scratch (C-layout -> A-layout)
  __shared__ u16b vt[8][32][72];     // per-wave V tile (32 keys x 64 dh, +8 pad)

  const int t = threadIdx.x;
  const int lane = t & 31, w = t >> 5;
  const int llo = lane & 15, lhi = lane >> 4;
  const int bh = blockIdx.y;
  const int b = bh >> 4, h = bh & 15;
  const int mtile = blockIdx.x * 8 + w;     // 0..63
  const int m0 = mtile * 16;

  // Q fragments for the two 32-wide K-steps over DH=64
  Frag qa[2];
  {
    const u16b* qrow = qb + ((size_t)(b * Sc + m0 + llo)) * Dc + h * 64;
#pragma unroll
    for (int kk = 0; kk < 2; ++kk) {
      qa[kk].u4[0] = *(const uint4*)(qrow + kk * 32 + lhi * 8);
      qa[kk].u4[1] = *(const uint4*)(qrow + kk * 32 + 16 + lhi * 8);
    }
  }

  float mr[8], lr[8];
  v8f o[4] = {};
#pragma unroll
  for (int r = 0; r < 8; ++r) { mr[r] = -1e30f; lr[r] = 0.0f; }

  for (int kt = 0; kt < Sc / 32; ++kt) {
    const int kbase = kt * 32;

    // stage V tile (32 keys x 64 dh) into this wave's LDS region
#if HAVE_TDM
    tdm_load_2d(lds_off(&vt[w][0][0]),
                vb + (size_t)(b * Sc + kbase) * Dc + h * 64,
                (unsigned)Dc, (unsigned)(Bc * Sc), 64u, 32u, (unsigned)Dc,
                4u, 3u); // pad 16B after each 128B row -> 72-short stride
#else
#pragma unroll
    for (int i = 0; i < 8; ++i) {
      int id = i * 32 + lane;
      int row = id >> 3, ch = id & 7;
      uint4 d = *(const uint4*)(vb + (size_t)(b * Sc + kbase + row) * Dc + h * 64 + ch * 8);
      *(uint4*)&vt[w][row][ch * 8] = d;
    }
#endif

    // scores S = Q(16x64) * K_t^T(64x32): two 16x16 key tiles
    v8f s0 = {}, s1 = {};
#pragma unroll
    for (int kk = 0; kk < 2; ++kk) {
      Frag kf0, kf1;
      const u16b* kr0 = kb + (size_t)(b * Sc + kbase + llo) * Dc + h * 64 + kk * 32;
      const u16b* kr1 = kb + (size_t)(b * Sc + kbase + 16 + llo) * Dc + h * 64 + kk * 32;
      kf0.u4[0] = ((const uint4*)kr0)[lhi]; kf0.u4[1] = ((const uint4*)kr0)[2 + lhi];
      kf1.u4[0] = ((const uint4*)kr1)[lhi]; kf1.u4[1] = ((const uint4*)kr1)[2 + lhi];
      s0 = __builtin_amdgcn_wmma_f32_16x16x32_bf16(false, qa[kk].v, false, kf0.v,
                                                   (short)0, s0, false, false);
      s1 = __builtin_amdgcn_wmma_f32_16x16x32_bf16(false, qa[kk].v, false, kf1.v,
                                                   (short)0, s1, false, false);
    }

    // key-padding mask (per lane = per key column)
    const float NEG = -__builtin_inff();
    if (mask[b * Sc + kbase + llo] == 0) {
#pragma unroll
      for (int r = 0; r < 8; ++r) s0[r] = NEG;
    }
    if (mask[b * Sc + kbase + 16 + llo] == 0) {
#pragma unroll
      for (int r = 0; r < 8; ++r) s1[r] = NEG;
    }

    // online softmax row statistics (rows live per (r, lane-half))
    float alpha[8];
#pragma unroll
    for (int r = 0; r < 8; ++r) {
      float tm = fmaxf(s0[r], s1[r]);
#pragma unroll
      for (int off = 1; off < 16; off <<= 1) tm = fmaxf(tm, __shfl_xor(tm, off, 32));
      float mn = fmaxf(mr[r], tm);
      alpha[r] = __expf(mr[r] - mn);
      mr[r] = mn;
      float p0 = __expf(s0[r] - mn);   // exp(-inf)=0 for masked
      float p1 = __expf(s1[r] - mn);
      s0[r] = p0; s1[r] = p1;
      float rs = p0 + p1;
#pragma unroll
      for (int off = 1; off < 16; off <<= 1) rs += __shfl_xor(rs, off, 32);
      lr[r] = lr[r] * alpha[r] + rs;
    }
#pragma unroll
    for (int j = 0; j < 4; ++j)
#pragma unroll
      for (int r = 0; r < 8; ++r) o[j][r] *= alpha[r];

    // P: C-layout -> LDS -> A-layout bf16 fragment
#pragma unroll
    for (int r = 0; r < 8; ++r) {
      int row = r + lhi * 8;
      pbuf[w][row][llo] = f2bf(s0[r]);
      pbuf[w][row][16 + llo] = f2bf(s1[r]);
    }
    wait_ds0(); // cross-lane DS exchange within the wave

    Frag pf;
    pf.u4[0] = *(const uint4*)&pbuf[w][llo][lhi * 8];
    pf.u4[1] = *(const uint4*)&pbuf[w][llo][16 + lhi * 8];

#if HAVE_TDM
    wait_tensor0(); // V tile DMA complete (overlapped with scores/softmax)
#endif

    // O(16x64) += P(16x32) * V_t(32x64): 4 n-tiles
#pragma unroll
    for (int j = 0; j < 4; ++j) {
      Frag vf;
#pragma unroll
      for (int e = 0; e < 16; ++e) vf.h[e] = vt[w][(e < 8 ? lhi * 8 + e : 16 + lhi * 8 + (e - 8))][j * 16 + llo];
      o[j] = __builtin_amdgcn_wmma_f32_16x16x32_bf16(false, pf.v, false, vf.v,
                                                     (short)0, o[j], false, false);
    }
  }

  // normalize and store bf16 output [b*S+s][h*64+dh]
#pragma unroll
  for (int j = 0; j < 4; ++j) {
#pragma unroll
    for (int r = 0; r < 8; ++r) {
      int row = m0 + r + lhi * 8;
      float denom = lr[r];
      ob[(size_t)(b * Sc + row) * Dc + h * 64 + j * 16 + llo] = f2bf(o[j][r] / denom);
    }
  }
}

// ---------------------------------------------------------------------------
// LayerNorm over last dim (D=1024), one row per 256-thread block
// ---------------------------------------------------------------------------
__global__ __launch_bounds__(256) void ln_kernel(const float* __restrict__ xin,
                                                 const float* __restrict__ g,
                                                 const float* __restrict__ bta,
                                                 float* __restrict__ xout) {
  __shared__ float rs[8], rss[8];
  const int row = blockIdx.x;
  const int t = threadIdx.x;
  float s = 0.0f, ss = 0.0f;
  for (int c = t; c < Dc; c += 256) {
    float v = xin[(size_t)row * Dc + c];
    s += v; ss += v * v;
  }
#pragma unroll
  for (int off = 16; off; off >>= 1) {
    s += __shfl_xor(s, off, 32);
    ss += __shfl_xor(ss, off, 32);
  }
  if ((t & 31) == 0) { rs[t >> 5] = s; rss[t >> 5] = ss; }
  __syncthreads();
  float tot = 0.0f, tots = 0.0f;
#pragma unroll
  for (int i = 0; i < 8; ++i) { tot += rs[i]; tots += rss[i]; }
  float mean = tot * (1.0f / Dc);
  float var = tots * (1.0f / Dc) - mean * mean;
  float inv = rsqrtf(var + 1e-5f);
  for (int c = t; c < Dc; c += 256) {
    float v = xin[(size_t)row * Dc + c];
    xout[(size_t)row * Dc + c] = (v - mean) * inv * g[c] + bta[c];
  }
}

// ---------------------------------------------------------------------------
// Host launcher
// ---------------------------------------------------------------------------
extern "C" void kernel_launch(void* const* d_in, const int* in_sizes, int n_in,
                              void* d_out, int out_size, void* d_ws, size_t ws_size,
                              hipStream_t stream) {
  (void)in_sizes; (void)n_in; (void)out_size; (void)ws_size;
  const int*   tokens = (const int*)d_in[0];
  const int*   mask   = (const int*)d_in[1];
  const float* emb    = (const float*)d_in[2];
  const float* Wq     = (const float*)d_in[3];
  const float* bq     = (const float*)d_in[4];
  const float* Wk     = (const float*)d_in[5];
  const float* bk     = (const float*)d_in[6];
  const float* Wv     = (const float*)d_in[7];
  const float* bv     = (const float*)d_in[8];
  const float* Wo     = (const float*)d_in[9];
  const float* bo     = (const float*)d_in[10];
  const float* ln1g   = (const float*)d_in[11];
  const float* ln1b   = (const float*)d_in[12];
  const float* W1     = (const float*)d_in[13];
  const float* b1     = (const float*)d_in[14];
  const float* W2     = (const float*)d_in[15];
  const float* b2     = (const float*)d_in[16];
  const float* ln2g   = (const float*)d_in[17];
  const float* ln2b   = (const float*)d_in[18];
  float* out = (float*)d_out;

  const size_t RD = (size_t)ROWS * Dc;          // 4M elements
  char* p = (char*)d_ws;
  float* x  = (float*)p; p += RD * 4;           // fp32 residual stream
  u16b* xb  = (u16b*)p;  p += RD * 2;           // bf16 activations
  u16b* qb_ = (u16b*)p;  p += RD * 2;
  u16b* kb_ = (u16b*)p;  p += RD * 2;
  u16b* vb_ = (u16b*)p;  p += RD * 2;
  u16b* ob_ = (u16b*)p;  p += RD * 2;
  u16b* h1b = (u16b*)p;  p += (size_t)ROWS * FFc * 2;  // 32MB
  u16b* wp  = (u16b*)p;  p += (size_t)FFc * Dc * 2;    // 8MB packed weight

  const int CONV_BLOCKS = (int)(RD / 256);      // 16384
  dim3 blk(256);

  embed_kernel<<<CONV_BLOCKS, blk, 0, stream>>>(tokens, emb, x);

  for (int l = 0; l < Lc; ++l) {
    const float* Wq_l = Wq + (size_t)l * Hc * Dc * DHc;
    const float* Wk_l = Wk + (size_t)l * Hc * Dc * DHc;
    const float* Wv_l = Wv + (size_t)l * Hc * Dc * DHc;
    const float* Wo_l = Wo + (size_t)l * Dc * Dc;
    const float* W1_l = W1 + (size_t)l * Dc * FFc;
    const float* W2_l = W2 + (size_t)l * FFc * Dc;

    convert_kernel<<<CONV_BLOCKS, blk, 0, stream>>>(x, xb);

    // Q (scaled by 1/sqrt(DH)), K, V projections
    pack_head_kernel<<<(Dc * Dc) / 256, blk, 0, stream>>>(Wq_l, wp);
    gemm_kernel<EPI_SCALE><<<dim3(Dc / 128, ROWS / 128), blk, 0, stream>>>(
        xb, wp, bq + l * Hc * DHc, (void*)qb_, Dc, Dc, 0.125f);
    pack_head_kernel<<<(Dc * Dc) / 256, blk, 0, stream>>>(Wk_l, wp);
    gemm_kernel<EPI_BF16><<<dim3(Dc / 128, ROWS / 128), blk, 0, stream>>>(
        xb, wp, bk + l * Hc * DHc, (void*)kb_, Dc, Dc, 1.0f);
    pack_head_kernel<<<(Dc * Dc) / 256, blk, 0, stream>>>(Wv_l, wp);
    gemm_kernel<EPI_BF16><<<dim3(Dc / 128, ROWS / 128), blk, 0, stream>>>(
        xb, wp, bv + l * Hc * DHc, (void*)vb_, Dc, Dc, 1.0f);

    // flash attention -> ob (bf16, head-concat layout)
    attention_kernel<<<dim3(Sc / 16 / 8, Bc * Hc), blk, 0, stream>>>(
        qb_, kb_, vb_, mask, ob_);

    // output projection, residual-accumulated into x
    pack_plain_kernel<<<(Dc * Dc) / 256, blk, 0, stream>>>(Wo_l, wp, Dc, Dc);
    gemm_kernel<EPI_RESID><<<dim3(Dc / 128, ROWS / 128), blk, 0, stream>>>(
        ob_, wp, bo + l * Dc, (void*)x, Dc, Dc, 1.0f);

    ln_kernel<<<ROWS, blk, 0, stream>>>(x, ln1g + l * Dc, ln1b + l * Dc, x);

    // FFN
    convert_kernel<<<CONV_BLOCKS, blk, 0, stream>>>(x, xb);
    pack_plain_kernel<<<(Dc * FFc) / 256, blk, 0, stream>>>(W1_l, wp, Dc, FFc);
    gemm_kernel<EPI_RELU><<<dim3(FFc / 128, ROWS / 128), blk, 0, stream>>>(
        xb, wp, b1 + l * FFc, (void*)h1b, FFc, Dc, 1.0f);
    pack_plain_kernel<<<(FFc * Dc) / 256, blk, 0, stream>>>(W2_l, wp, FFc, Dc);
    gemm_kernel<EPI_RESID><<<dim3(Dc / 128, ROWS / 128), blk, 0, stream>>>(
        h1b, wp, b2 + l * Dc, (void*)x, Dc, FFc, 1.0f);

    ln_kernel<<<ROWS, blk, 0, stream>>>(x, ln2g + l * Dc, ln2b + l * Dc,
                                        (l == Lc - 1) ? out : x);
  }
}
